// EconomicSelfAttention_67138928771456
// MI455X (gfx1250) — matrix operand order repaired and checked by reading, not compiled, not run on hardware
//
#include <hip/hip_runtime.h>
#include <hip/hip_bf16.h>

#define B_ 2
#define S_ 4096
#define D_ 512
#define H_ 8
#define DH_ 64

typedef __attribute__((ext_vector_type(16))) _Float16 v16h;
typedef __attribute__((ext_vector_type(8)))  _Float16 v8h;
typedef __attribute__((ext_vector_type(8)))  float    v8f;

static __device__ __forceinline__ v8f wmma_f16(v16h a, v16h b, v8f c) {
  return __builtin_amdgcn_wmma_f32_16x16x32_f16(false, a, false, b, (short)0, c, false, false);
}

// A operand, 16x32 fp16: lane L holds row m=L&15; elems 0..7 = K(koff..koff+7),
// elems 8..15 = K(16+koff..16+koff+7), koff = (L>>4)*8.  (ISA 05_wmma 16-bit A layout)
static __device__ __forceinline__ v16h load_a16(const _Float16* base, int ld) {
  const int lane = threadIdx.x & 31;
  const int m = lane & 15;
  const int koff = (lane >> 4) << 3;
  const _Float16* p = base + m * ld + koff;
  v8h lo = *(const v8h*)(p);
  v8h hi = *(const v8h*)(p + 16);
  v16h r;
#pragma unroll
  for (int j = 0; j < 8; ++j) { r[j] = lo[j]; r[j + 8] = hi[j]; }
  return r;
}

// B operand, 32x16 fp16 where memory holds Bt[n][k] row-major (B[k][n] = Bt[n][k]):
// lane L = column n=L&15; elems 0..7 = K(koff..koff+7), elems 8..15 = K(16+koff..16+koff+7), koff = (L>>4)*8.
static __device__ __forceinline__ v16h load_b16_rows(const _Float16* base, int ld) {
  const int lane = threadIdx.x & 31;
  const int n = lane & 15;
  const int kb = (lane >> 4) << 3;
  const _Float16* p = base + n * ld + kb;
  v8h lo = *(const v8h*)(p);
  v8h hi = *(const v8h*)(p + 16);
  v16h r;
#pragma unroll
  for (int j = 0; j < 8; ++j) { r[j] = lo[j]; r[j + 8] = hi[j]; }
  return r;
}

// ---------------- fp32 -> fp16 convert ----------------
__global__ void cvt_f32_f16_kernel(const float* __restrict__ in,
                                   _Float16* __restrict__ out, int n) {
  int i = blockIdx.x * blockDim.x + threadIdx.x;
  if (i < n) out[i] = (_Float16)in[i];
}

// ---------------- Q/K projection: C[m,n] = sum_k X[m,k]*W[n,k] + b[n] ----------------
// One wave computes a 16x64 tile. Output stored head-major [B,H,S,DH] fp16.
__global__ void proj_qk_kernel(const _Float16* __restrict__ X,   // [B*S, D]
                               const _Float16* __restrict__ W,   // [D, D] (row = out feature)
                               const float* __restrict__ bias,   // [D]
                               _Float16* __restrict__ out) {     // [B,H,S,DH]
  const int lane = threadIdx.x;
  const int n0 = blockIdx.x * 64;
  const int m0 = blockIdx.y * 16;

  v8f acc[4] = {};
  for (int k0 = 0; k0 < D_; k0 += 32) {
    v16h a = load_a16(X + (size_t)m0 * D_ + k0, D_);
#pragma unroll
    for (int i = 0; i < 4; ++i) {
      v16h b = load_b16_rows(W + (size_t)(n0 + 16 * i) * D_ + k0, D_);
      acc[i] = wmma_f16(a, b, acc[i]);
    }
  }
  const int nloc = lane & 15;
  const int mbase = (lane >> 4) << 3;
#pragma unroll
  for (int i = 0; i < 4; ++i) {
    const int n = n0 + 16 * i + nloc;
    const int h = n / DH_, d = n % DH_;
    const float bv = bias[n];
#pragma unroll
    for (int r = 0; r < 8; ++r) {
      const int mg = m0 + mbase + r;
      const int b = mg / S_, s = mg % S_;
      out[(((size_t)b * H_ + h) * S_ + s) * DH_ + d] = (_Float16)(acc[i][r] + bv);
    }
  }
}

// ---------------- V projection: same GEMM, but store TRANSPOSED [B,H,DH,S] ----------------
// This makes the attention AV B-operand (B[t][d] = V^T[d][t]) contiguous in memory.
__global__ void proj_v_kernel(const _Float16* __restrict__ X,    // [B*S, D]
                              const _Float16* __restrict__ W,    // [D, D]
                              const float* __restrict__ bias,    // [D]
                              _Float16* __restrict__ out) {      // [B,H,DH,S]
  const int lane = threadIdx.x;
  const int n0 = blockIdx.x * 64;
  const int m0 = blockIdx.y * 16;

  v8f acc[4] = {};
  for (int k0 = 0; k0 < D_; k0 += 32) {
    v16h a = load_a16(X + (size_t)m0 * D_ + k0, D_);
#pragma unroll
    for (int i = 0; i < 4; ++i) {
      v16h b = load_b16_rows(W + (size_t)(n0 + 16 * i) * D_ + k0, D_);
      acc[i] = wmma_f16(a, b, acc[i]);
    }
  }
  const int nloc = lane & 15;
  const int mbase = (lane >> 4) << 3;
#pragma unroll
  for (int i = 0; i < 4; ++i) {
    const int n = n0 + 16 * i + nloc;
    const int h = n / DH_, d = n % DH_;
    const float bv = bias[n];
#pragma unroll
    for (int r = 0; r < 8; ++r) {
      const int mg = m0 + mbase + r;
      const int b = mg / S_, s = mg % S_;
      out[(((size_t)b * H_ + h) * DH_ + d) * S_ + s] = (_Float16)(acc[i][r] + bv);
    }
  }
}

// ---------------- Flash attention with hyperbolic positional bias ----------------
// One wave handles 32 query rows (two 16-row sub-tiles sharing K/V^T fragments)
// for one (b,h). t processed in chunks of 32; the last chunk ends exactly at the
// diagonal block, so no out-of-range t reads occur.
__global__ void attn_kernel(const _Float16* __restrict__ Q,    // [B,H,S,DH]
                            const _Float16* __restrict__ K,    // [B,H,S,DH]
                            const _Float16* __restrict__ Vt,   // [B,H,DH,S]
                            const float* __restrict__ gamma,   // [B,S]
                            _Float16* __restrict__ O) {        // [B,S,D] heads concatenated
  __shared__ _Float16 ldsP[2][16 * 48];  // two 16x32 P tiles, rows padded to 48 halfs
  const int lane = threadIdx.x;
  const int bh = blockIdx.y;
  const int b = bh / H_, h = bh % H_;
  const int s0 = blockIdx.x * 32;

  const _Float16* Qh  = Q  + (size_t)bh * S_ * DH_;
  const _Float16* Kh  = K  + (size_t)bh * S_ * DH_;
  const _Float16* Vth = Vt + (size_t)bh * DH_ * S_;

  const int nloc = lane & 15;
  const int mbase = (lane >> 4) << 3;
  const int kb = (lane >> 4) << 3;

  // Q fragments (reused across all t chunks): [row sub-tile][d chunk]
  v16h qa[2][2];
#pragma unroll
  for (int u = 0; u < 2; ++u) {
    qa[u][0] = load_a16(Qh + (size_t)(s0 + 16 * u) * DH_ + 0, DH_);
    qa[u][1] = load_a16(Qh + (size_t)(s0 + 16 * u) * DH_ + 32, DH_);
  }

  float grow[2][8], row_m[2][8], row_l[2][8];
#pragma unroll
  for (int u = 0; u < 2; ++u)
#pragma unroll
    for (int r = 0; r < 8; ++r) {
      grow[u][r] = gamma[(size_t)b * S_ + s0 + 16 * u + mbase + r];
      row_m[u][r] = -1e30f;
      row_l[u][r] = 0.f;
    }
  v8f o[2][4] = {};

  for (int t0 = 0; t0 <= s0 + 31; t0 += 32) {
    // prefetch next chunk's K rows and V^T columns (global_prefetch_b8)
    const int tn = t0 + 32;
    if (tn <= s0 + 31) {
      __builtin_prefetch(Kh + (size_t)(tn + lane) * DH_, 0, 0);
      __builtin_prefetch(Vth + (size_t)lane * S_ + tn, 0, 0);
      __builtin_prefetch(Vth + (size_t)(lane + 32) * S_ + tn, 0, 0);
    }
    // ---- K B-operand fragments for t0..t0+31, shared by both row sub-tiles ----
    v16h kf[2][2];  // [t sub-tile][d chunk]
#pragma unroll
    for (int sub = 0; sub < 2; ++sub) {
      const _Float16* p0 = Kh + (size_t)(t0 + 16 * sub + nloc) * DH_;
      v8h a0 = *(const v8h*)(p0 + kb);
      v8h a1 = *(const v8h*)(p0 + kb + 16);
      v8h a2 = *(const v8h*)(p0 + 32 + kb);
      v8h a3 = *(const v8h*)(p0 + 32 + kb + 16);
#pragma unroll
      for (int j = 0; j < 8; ++j) {
        kf[sub][0][j] = a0[j]; kf[sub][0][j + 8] = a1[j];
        kf[sub][1][j] = a2[j]; kf[sub][1][j + 8] = a3[j];
      }
    }
    __syncthreads();  // previous iteration's ldsP reads complete before rewrite
#pragma unroll
    for (int u = 0; u < 2; ++u) {
      // ---- scores for sub-tile u: two 16x16 tiles covering t0..t0+31 ----
      v8f sc0 = {}, sc1 = {};
      sc0 = wmma_f16(qa[u][0], kf[0][0], sc0);
      sc0 = wmma_f16(qa[u][1], kf[0][1], sc0);
      sc1 = wmma_f16(qa[u][0], kf[1][0], sc1);
      sc1 = wmma_f16(qa[u][1], kf[1][1], sc1);
      // ---- bias + causal mask + online softmax ----
      float p0[8], p1[8];
#pragma unroll
      for (int r = 0; r < 8; ++r) {
        const int s = s0 + 16 * u + mbase + r;
        const int ta = t0 + nloc;
        const int tb = t0 + 16 + nloc;
        p0[r] = (ta > s) ? -1e30f
                : sc0[r] * 0.125f - log1pf(grow[u][r] * (float)(s - ta) + 1e-8f);
        p1[r] = (tb > s) ? -1e30f
                : sc1[r] * 0.125f - log1pf(grow[u][r] * (float)(s - tb) + 1e-8f);
      }
#pragma unroll
      for (int r = 0; r < 8; ++r) {
        float mx = fmaxf(p0[r], p1[r]);
#pragma unroll
        for (int off = 1; off < 16; off <<= 1) mx = fmaxf(mx, __shfl_xor(mx, off, 32));
        const float mnew = fmaxf(row_m[u][r], mx);
        const float scale = __expf(row_m[u][r] - mnew);
        row_m[u][r] = mnew;
        p0[r] = __expf(p0[r] - mnew);
        p1[r] = __expf(p1[r] - mnew);
        float rs = p0[r] + p1[r];
#pragma unroll
        for (int off = 1; off < 16; off <<= 1) rs += __shfl_xor(rs, off, 32);
        row_l[u][r] = row_l[u][r] * scale + rs;
#pragma unroll
        for (int i = 0; i < 4; ++i) o[u][i][r] *= scale;
      }
      // ---- P: C-layout -> A-layout staging ----
#pragma unroll
      for (int r = 0; r < 8; ++r) {
        ldsP[u][(mbase + r) * 48 + nloc] = (_Float16)p0[r];
        ldsP[u][(mbase + r) * 48 + 16 + nloc] = (_Float16)p1[r];
      }
    }
    __syncthreads();
    const v16h pa0 = load_a16(ldsP[0], 48);
    const v16h pa1 = load_a16(ldsP[1], 48);
    // ---- AV: two 16x64 out tiles, K-dim = 32 t values; V^T fragments shared ----
#pragma unroll
    for (int i = 0; i < 4; ++i) {
      v16h vb = load_b16_rows(Vth + (size_t)(16 * i) * S_ + t0, S_);
      o[0][i] = wmma_f16(pa0, vb, o[0][i]);
      o[1][i] = wmma_f16(pa1, vb, o[1][i]);
    }
  }
  // ---- normalize and store (concat heads) ----
#pragma unroll
  for (int u = 0; u < 2; ++u)
#pragma unroll
    for (int i = 0; i < 4; ++i) {
      const int d = 16 * i + nloc;
#pragma unroll
      for (int r = 0; r < 8; ++r) {
        const int s = s0 + 16 * u + mbase + r;
        const float val = o[u][i][r] / row_l[u][r];
        O[((size_t)b * S_ + s) * D_ + h * DH_ + d] = (_Float16)val;
      }
    }
}

// ---------------- output projection: fp32 result + bias -> d_out ----------------
__global__ void out_proj_kernel(const _Float16* __restrict__ X,  // [B*S, D] fp16
                                const _Float16* __restrict__ W,  // [D, D]
                                const float* __restrict__ bias,  // [D]
                                float* __restrict__ out) {       // [B*S, D] fp32
  const int lane = threadIdx.x;
  const int n0 = blockIdx.x * 64;
  const int m0 = blockIdx.y * 16;

  v8f acc[4] = {};
  for (int k0 = 0; k0 < D_; k0 += 32) {
    v16h a = load_a16(X + (size_t)m0 * D_ + k0, D_);
#pragma unroll
    for (int i = 0; i < 4; ++i) {
      v16h b = load_b16_rows(W + (size_t)(n0 + 16 * i) * D_ + k0, D_);
      acc[i] = wmma_f16(a, b, acc[i]);
    }
  }
  const int nloc = lane & 15;
  const int mbase = (lane >> 4) << 3;
#pragma unroll
  for (int i = 0; i < 4; ++i) {
    const int n = n0 + 16 * i + nloc;
    const float bv = bias[n];
#pragma unroll
    for (int r = 0; r < 8; ++r) {
      const int mg = m0 + mbase + r;
      out[(size_t)mg * D_ + n] = acc[i][r] + bv;
    }
  }
}

extern "C" void kernel_launch(void* const* d_in, const int* in_sizes, int n_in,
                              void* d_out, int out_size, void* d_ws, size_t ws_size,
                              hipStream_t stream) {
  (void)in_sizes; (void)n_in; (void)out_size; (void)ws_size;
  const float* x     = (const float*)d_in[0];
  const float* gamma = (const float*)d_in[1];
  const float* Wq    = (const float*)d_in[2];
  const float* bq    = (const float*)d_in[3];
  const float* Wk    = (const float*)d_in[4];
  const float* bk    = (const float*)d_in[5];
  const float* Wv    = (const float*)d_in[6];
  const float* bv    = (const float*)d_in[7];
  const float* Wo    = (const float*)d_in[8];
  const float* bo    = (const float*)d_in[9];
  float* out = (float*)d_out;

  const size_t MK = (size_t)B_ * S_ * D_;  // 4M elements
  const size_t WW = (size_t)D_ * D_;       // 256K elements
  _Float16* ws  = (_Float16*)d_ws;
  _Float16* xh  = ws;
  _Float16* wqh = xh + MK;
  _Float16* wkh = wqh + WW;
  _Float16* wvh = wkh + WW;
  _Float16* woh = wvh + WW;
  _Float16* Qh  = woh + WW;
  _Float16* Kh  = Qh + MK;
  _Float16* Vth = Kh + MK;
  _Float16* Ah  = Vth + MK;

  cvt_f32_f16_kernel<<<(int)((MK + 255) / 256), 256, 0, stream>>>(x, xh, (int)MK);
  cvt_f32_f16_kernel<<<(int)((WW + 255) / 256), 256, 0, stream>>>(Wq, wqh, (int)WW);
  cvt_f32_f16_kernel<<<(int)((WW + 255) / 256), 256, 0, stream>>>(Wk, wkh, (int)WW);
  cvt_f32_f16_kernel<<<(int)((WW + 255) / 256), 256, 0, stream>>>(Wv, wvh, (int)WW);
  cvt_f32_f16_kernel<<<(int)((WW + 255) / 256), 256, 0, stream>>>(Wo, woh, (int)WW);

  dim3 gproj(D_ / 64, (B_ * S_) / 16);  // (8, 512)
  proj_qk_kernel<<<gproj, 32, 0, stream>>>(xh, wqh, bq, Qh);
  proj_qk_kernel<<<gproj, 32, 0, stream>>>(xh, wkh, bk, Kh);
  proj_v_kernel <<<gproj, 32, 0, stream>>>(xh, wvh, bv, Vth);

  attn_kernel<<<dim3(S_ / 32, B_ * H_), 32, 0, stream>>>(Qh, Kh, Vth, gamma, Ah);

  out_proj_kernel<<<gproj, 32, 0, stream>>>(Ah, woh, bo, out);
}
